// Fuse_line_41223096107479
// MI455X (gfx1250) — matrix operand order repaired and checked by reading, not compiled, not run on hardware
//
#include <hip/hip_runtime.h>
#include <hip/hip_bf16.h>

// ---------------------------------------------------------------------------
// Types for CDNA5 WMMA (gfx1250, wave32)
// ---------------------------------------------------------------------------
typedef __attribute__((ext_vector_type(16))) __bf16 v16bf;
typedef __attribute__((ext_vector_type(8)))  __bf16 v8bf;
typedef __attribute__((ext_vector_type(8)))  float  v8f;

union ABFrag { v16bf v; v8bf h[2]; };

__device__ __forceinline__ __bf16 f2bf(float f) {
    unsigned int u = __float_as_uint(f);
    unsigned int r = (u + 0x7FFFu + ((u >> 16) & 1u)) >> 16;
    unsigned short s = (unsigned short)r;
    return __builtin_bit_cast(__bf16, s);
}

// ---- CDNA5 async global->LDS path (guarded; falls back to reg staging) ----
#if defined(__gfx1250__) && __has_builtin(__builtin_amdgcn_global_load_async_to_lds_b128)
#define ASYNC_LDS 1
typedef int v4i_vec __attribute__((vector_size(16)));
typedef __attribute__((address_space(1))) v4i_vec* as1_v4i;
typedef __attribute__((address_space(3))) v4i_vec* as3_v4i;
__device__ __forceinline__ void async_ld16(const void* g, void* l) {
    __builtin_amdgcn_global_load_async_to_lds_b128(
        (as1_v4i)(v4i_vec*)const_cast<void*>(g),
        (as3_v4i)(v4i_vec*)l, 0, 0);
}
__device__ __forceinline__ void async_wait() {
#if __has_builtin(__builtin_amdgcn_s_wait_asynccnt)
    __builtin_amdgcn_s_wait_asynccnt(0);
#else
    asm volatile("s_wait_asynccnt 0x0" ::: "memory");
#endif
}
#else
#define ASYNC_LDS 0
__device__ __forceinline__ void async_wait() {}
#endif

// ---------------------------------------------------------------------------
// Generic bf16 WMMA GEMM (compile-time TA/TB):
//   D[M,N] = alpha * A'·B' (+ bias[m]) (+ add[m,n])
//   A' = TA ? A(stored [K,M])^T : A(stored [M,K])
//   B' = TB ? B(stored [N,K])^T : B(stored [K,N])
// Workgroup tile 128x128, 256 threads = 8 waves, wave tile 64x32,
// K stepped by 32, double-buffered LDS, async/LDS-staged prefetch.
// ---------------------------------------------------------------------------
template<int TA, int TB>
__global__ __launch_bounds__(256)
void gemm_k(const __bf16* __restrict__ A, const __bf16* __restrict__ Bm,
            float* __restrict__ outF, __bf16* __restrict__ outB,
            const float* __restrict__ bias, const float* __restrict__ addF,
            const float* __restrict__ alpha,
            int M, int N, int K)
{
    __shared__ __align__(16) __bf16 ldsA[2][128][40];   // [buf][m][k], 80B rows
    __shared__ __align__(16) __bf16 ldsBT[2][128][40];  // [buf][n][k]

    const int tid  = threadIdx.x;
    const int lane = tid & 31;
    const int w    = tid >> 5;
    const int wr   = w >> 2;         // 0..1 -> 64-row slab
    const int wc   = w & 3;          // 0..3 -> 32-col slab
    const int tiM  = blockIdx.y * 128;
    const int tiN  = blockIdx.x * 128;

    // ---- strength-reduced per-thread staging state -------------------------
    // A side
    const __bf16 *pA0 = nullptr, *pA1 = nullptr, *pAs = nullptr;
    int mA0 = 0, kcA0 = 0, sAm = 0, sAkb = 0;
    size_t twoM = 0;
    if constexpr (TA == 0) {
        mA0 = tid >> 2; kcA0 = (tid & 3) * 8;        // chunk rows mA0, mA0+64
        pA0 = A + (size_t)(tiM + mA0) * K + kcA0;
        pA1 = pA0 + (size_t)64 * K;
    } else {
        sAm = tid & 127; sAkb = tid >> 7;            // k = sAkb + 2i
        pAs = A + (size_t)sAkb * M + tiM + sAm;
        twoM = (size_t)2 * M;
    }
    // B side
    const __bf16 *pB0 = nullptr, *pB1 = nullptr, *pBs = nullptr;
    int nB0 = 0, kcB0 = 0, sBn = 0, sBkb = 0;
    size_t twoN = 0;
    if constexpr (TB == 1) {
        nB0 = tid >> 2; kcB0 = (tid & 3) * 8;
        pB0 = Bm + (size_t)(tiN + nB0) * K + kcB0;
        pB1 = pB0 + (size_t)64 * K;
    } else {
        sBn = tid & 127; sBkb = tid >> 7;
        pBs = Bm + (size_t)sBkb * N + tiN + sBn;
        twoN = (size_t)2 * N;
    }

    __bf16 ra[16], rb[16];
#if !ASYNC_LDS
    v8bf ca0, ca1, cb0, cb1;
#endif

    auto issueA = [&](int buf) {
        if constexpr (TA == 0) {
#if ASYNC_LDS
            async_ld16(pA0, &ldsA[buf][mA0][kcA0]);
            async_ld16(pA1, &ldsA[buf][mA0 + 64][kcA0]);
#else
            ca0 = *(const v8bf*)pA0;
            ca1 = *(const v8bf*)pA1;
#endif
            pA0 += 32; pA1 += 32;
        } else {
            const __bf16* p = pAs;
#pragma unroll
            for (int i = 0; i < 16; ++i) { ra[i] = *p; p += twoM; }
            pAs += (size_t)16 * twoM;
        }
    };
    auto commitA = [&](int buf) {
        if constexpr (TA == 0) {
#if !ASYNC_LDS
            *(v8bf*)&ldsA[buf][mA0][kcA0]      = ca0;
            *(v8bf*)&ldsA[buf][mA0 + 64][kcA0] = ca1;
#endif
        } else {
#pragma unroll
            for (int i = 0; i < 16; ++i) ldsA[buf][sAm][sAkb + 2 * i] = ra[i];
        }
    };
    auto issueB = [&](int buf) {
        if constexpr (TB == 1) {
#if ASYNC_LDS
            async_ld16(pB0, &ldsBT[buf][nB0][kcB0]);
            async_ld16(pB1, &ldsBT[buf][nB0 + 64][kcB0]);
#else
            cb0 = *(const v8bf*)pB0;
            cb1 = *(const v8bf*)pB1;
#endif
            pB0 += 32; pB1 += 32;
        } else {
            const __bf16* p = pBs;
#pragma unroll
            for (int i = 0; i < 16; ++i) { rb[i] = *p; p += twoN; }
            pBs += (size_t)16 * twoN;
        }
    };
    auto commitB = [&](int buf) {
        if constexpr (TB == 1) {
#if !ASYNC_LDS
            *(v8bf*)&ldsBT[buf][nB0][kcB0]      = cb0;
            *(v8bf*)&ldsBT[buf][nB0 + 64][kcB0] = cb1;
#endif
        } else {
#pragma unroll
            for (int i = 0; i < 16; ++i) ldsBT[buf][sBn][sBkb + 2 * i] = rb[i];
        }
    };

    v8f acc[4][2];
#pragma unroll
    for (int mf = 0; mf < 4; ++mf)
#pragma unroll
        for (int nf = 0; nf < 2; ++nf)
            acc[mf][nf] = v8f{0.f,0.f,0.f,0.f,0.f,0.f,0.f,0.f};

    const int nsteps = K >> 5;

    // prologue: stage tile 0 into buffer 0
    issueA(0); issueB(0);
    commitA(0); commitB(0);
    if constexpr (TA == 0 || TB == 1) async_wait();
    __syncthreads();

    const int mrow = lane & 15;
    const int koA  = (lane < 16) ? 0 : 8;    // A frag: K {0..7,16..23}/{8..15,24..31}
    const int koB  = (lane < 16) ? 0 : 16;   // unused; B frag reads K via koA, same order as A frag

    for (int t = 0; t < nsteps; ++t) {
        const int cur = t & 1, nxt = cur ^ 1;
        const bool hasNext = (t + 1 < nsteps);
        if (hasNext) { issueA(nxt); issueB(nxt); }

        ABFrag afr[4];
        v16bf  bfr[2];
#pragma unroll
        for (int mf = 0; mf < 4; ++mf) {
            int mm = wr * 64 + mf * 16 + mrow;
            afr[mf].h[0] = *(const v8bf*)&ldsA[cur][mm][koA];
            afr[mf].h[1] = *(const v8bf*)&ldsA[cur][mm][koA + 16];
        }
#pragma unroll
        for (int nf = 0; nf < 2; ++nf) {
            int nn = wc * 32 + nf * 16 + mrow;
            bfr[nf] = __builtin_shufflevector(*(const v8bf*)&ldsBT[cur][nn][koA], *(const v8bf*)&ldsBT[cur][nn][koA + 16], 0,1,2,3,4,5,6,7,8,9,10,11,12,13,14,15);
        }
#pragma unroll
        for (int mf = 0; mf < 4; ++mf)
#pragma unroll
            for (int nf = 0; nf < 2; ++nf)
                acc[mf][nf] = __builtin_amdgcn_wmma_f32_16x16x32_bf16(
                    false, afr[mf].v, false, bfr[nf], (short)0, acc[mf][nf],
                    false, false);

        if (hasNext) {
            commitA(nxt); commitB(nxt);
            if constexpr (TA == 0 || TB == 1) async_wait();
        }
        __syncthreads();
    }

    // ---- epilogue
    const float al = alpha ? alpha[0] : 1.0f;
    const int mo = (lane < 16) ? 0 : 8;      // C/D layout: row = r + 8*(lane>=16)
    const int nl = lane & 15;
#pragma unroll
    for (int mf = 0; mf < 4; ++mf)
#pragma unroll
        for (int nf = 0; nf < 2; ++nf) {
            int mb = tiM + wr * 64 + mf * 16 + mo;
            int nb = tiN + wc * 32 + nf * 16 + nl;
#pragma unroll
            for (int r = 0; r < 8; ++r) {
                int m = mb + r;
                float v = acc[mf][nf][r] * al;
                if (bias) v += bias[m];
                size_t o = (size_t)m * N + nb;
                if (addF) v += addF[o];
                if (outF) outF[o] = v;
                if (outB) outB[o] = f2bf(v);
            }
        }
}

// ---------------------------------------------------------------------------
// Row softmax: one workgroup per row, f32 in -> bf16 probabilities out
// ---------------------------------------------------------------------------
__global__ __launch_bounds__(256)
void softmax_rows(const float* __restrict__ S, __bf16* __restrict__ P, int ncols)
{
    const int row = blockIdx.x;
    const float* s = S + (size_t)row * ncols;
    __bf16* p = P + (size_t)row * ncols;
    __shared__ float red[256];

    float mx = -3.4e38f;
    for (int i = threadIdx.x; i < ncols; i += 256) mx = fmaxf(mx, s[i]);
    red[threadIdx.x] = mx; __syncthreads();
    for (int st = 128; st > 0; st >>= 1) {
        if (threadIdx.x < st)
            red[threadIdx.x] = fmaxf(red[threadIdx.x], red[threadIdx.x + st]);
        __syncthreads();
    }
    mx = red[0]; __syncthreads();

    float sum = 0.f;
    for (int i = threadIdx.x; i < ncols; i += 256) sum += __expf(s[i] - mx);
    red[threadIdx.x] = sum; __syncthreads();
    for (int st = 128; st > 0; st >>= 1) {
        if (threadIdx.x < st) red[threadIdx.x] += red[threadIdx.x + st];
        __syncthreads();
    }
    float inv = 1.0f / red[0];
    for (int i = threadIdx.x; i < ncols; i += 256)
        p[i] = f2bf(__expf(s[i] - mx) * inv);
}

// ---------------------------------------------------------------------------
// Small helpers
// ---------------------------------------------------------------------------
__global__ void cvt_bf16(const float* __restrict__ in, __bf16* __restrict__ out, int n)
{
    int i = blockIdx.x * 256 + threadIdx.x;
    if (i < n) out[i] = f2bf(in[i]);
}

// out[c,j] = sum_i W[c,i] * in[i,j]   (J = 36 or 9, K = 256)
__global__ void small_conv(const float* __restrict__ W, const float* __restrict__ in,
                           float* __restrict__ out, int Cc, int J)
{
    int idx = blockIdx.x * 256 + threadIdx.x;
    if (idx >= Cc * J) return;
    int c = idx / J, j = idx % J;
    float s = 0.f;
    for (int i = 0; i < Cc; ++i) s += W[(size_t)c * Cc + i] * in[(size_t)i * J + j];
    out[idx] = s;
}

// ml[d,j] = sum_p mv[d,p]*Wfc[j,p] + bfc[j]
__global__ void mline_k(const float* __restrict__ mv, const float* __restrict__ Wfc,
                        const float* __restrict__ bfc, float* __restrict__ ml)
{
    int idx = blockIdx.x * 256 + threadIdx.x;
    if (idx >= 256 * 36) return;
    int d = idx / 36, j = idx % 36;
    float s = bfc[j];
#pragma unroll
    for (int p = 0; p < 9; ++p) s += mv[d * 9 + p] * Wfc[j * 9 + p];
    ml[idx] = s;
}

// fm[c,d] = sum_j zv[c,j]*ml[d,j]  (f32 + bf16 outputs)
__global__ void fusemz_k(const float* __restrict__ zv, const float* __restrict__ ml,
                         float* __restrict__ fm, __bf16* __restrict__ fmb)
{
    int idx = blockIdx.x * 256 + threadIdx.x;
    if (idx >= 256 * 256) return;
    int c = idx >> 8, d = idx & 255;
    float s = 0.f;
#pragma unroll
    for (int j = 0; j < 36; ++j) s += zv[c * 36 + j] * ml[d * 36 + j];
    fm[idx] = s;
    fmb[idx] = f2bf(s);
}

// final: sigmoid -> softmax over {5,6} -> gated sum + residual
// o1T/o2T are [N,C] (transposed), already scaled by gamma/gamma2.
__global__ void final_k(const float* __restrict__ H5, const float* __restrict__ H6,
                        const float* __restrict__ o1T, const float* __restrict__ o2T,
                        const float* __restrict__ x, float* __restrict__ out,
                        int Cc, int Nn)
{
    int idx = blockIdx.x * 256 + threadIdx.x;
    if (idx >= Cc * Nn) return;
    int c = idx / Nn, n = idx % Nn;
    float s5 = 1.f / (1.f + __expf(-H5[idx]));
    float s6 = 1.f / (1.f + __expf(-H6[idx]));
    float e5 = __expf(s5), e6 = __expf(s6);
    float w5 = e5 / (e5 + e6);
    float o1 = o1T[(size_t)n * Cc + c];
    float o2 = o2T[(size_t)n * Cc + c];
    out[idx] = w5 * o1 + (1.f - w5) * o2 + x[idx];
}

// ---------------------------------------------------------------------------
// Host launcher
// ---------------------------------------------------------------------------
extern "C" void kernel_launch(void* const* d_in, const int* in_sizes, int n_in,
                              void* d_out, int out_size, void* d_ws, size_t ws_size,
                              hipStream_t stream)
{
    (void)in_sizes; (void)n_in; (void)out_size; (void)ws_size;
    constexpr int C = 256, N = 4096, NB = 4;
    const size_t CN = (size_t)C * N;
    const size_t NN = (size_t)N * N;

    const float* x    = (const float*)d_in[0];
    const float* y    = (const float*)d_in[1];
    const float* z    = (const float*)d_in[2];
    const float* m    = (const float*)d_in[3];
    const float* Wv   = (const float*)d_in[4];
    const float* Wv2  = (const float*)d_in[5];
    const float* Wq   = (const float*)d_in[6];
    const float* bq   = (const float*)d_in[7];
    const float* Wk   = (const float*)d_in[8];
    const float* bk   = (const float*)d_in[9];
    const float* Wq2  = (const float*)d_in[10];
    const float* bq2  = (const float*)d_in[11];
    const float* Wk2  = (const float*)d_in[12];
    const float* bk2  = (const float*)d_in[13];
    const float* Win1 = (const float*)d_in[14];
    const float* bin1 = (const float*)d_in[15];
    const float* Win3 = (const float*)d_in[16];
    const float* bin3 = (const float*)d_in[17];
    const float* Wtr  = (const float*)d_in[18];
    const float* btr  = (const float*)d_in[19];
    const float* Wout2 = (const float*)d_in[20];
    const float* bout2 = (const float*)d_in[21];
    const float* Wout3 = (const float*)d_in[22];
    const float* bout3 = (const float*)d_in[23];
    const float* Wfc  = (const float*)d_in[24];
    const float* bfc  = (const float*)d_in[25];
    const float* gamma  = (const float*)d_in[26];
    const float* gamma2 = (const float*)d_in[27];
    float* out = (float*)d_out;

    // ---- workspace bump allocator
    size_t off = 0;
    auto alloc = [&](size_t bytes) -> char* {
        char* p = (char*)d_ws + off;
        off += (bytes + 255) & ~(size_t)255;
        return p;
    };
    auto allocB = [&](size_t elems) { return (__bf16*)alloc(elems * 2); };
    auto allocF = [&](size_t elems) { return (float*)alloc(elems * 4); };

    // persistent bf16 copies
    __bf16* Wv_b   = allocB(C * C);
    __bf16* Wv2_b  = allocB(C * C);
    __bf16* Wq_b   = allocB(C * C);
    __bf16* Wk_b   = allocB(C * C);
    __bf16* Wq2_b  = allocB(C * C);
    __bf16* Wk2_b  = allocB(C * C);
    __bf16* Win1_b = allocB(C * C);
    __bf16* Win3_b = allocB(C * C);
    __bf16* Wo2_b  = allocB(C * C);
    __bf16* Wo3_b  = allocB(C * C);
    __bf16* Wtr_b  = allocB((size_t)(2 * C) * (2 * C));
    __bf16* x_b    = allocB((size_t)NB * CN);
    __bf16* y_b    = allocB((size_t)NB * CN);

    // per-batch scratch (reused)
    float*  yv     = allocF(CN);
    __bf16* yv_b   = allocB(CN);
    float*  zv     = allocF(C * 36);
    float*  mv     = allocF(C * 9);
    float*  ml     = allocF(C * 36);
    float*  fm     = allocF(C * C);
    __bf16* fm_b   = allocB(C * C);
    float*  fmzy   = allocF(CN);
    __bf16* fmzy_b = allocB(CN);
    __bf16* q_bf   = allocB(CN);
    __bf16* k_bf   = allocB(CN);
    __bf16* q2_bf  = allocB(CN);
    __bf16* k2_bf  = allocB(CN);
    __bf16* v2_bf  = allocB(CN);
    float*  S      = allocF(NN);
    __bf16* P      = allocB(NN);
    float*  o1T    = allocF(CN);
    __bf16* o1T_b  = allocB(CN);
    float*  o2T    = allocF(CN);
    __bf16* o2T_b  = allocB(CN);
    __bf16* Hcat   = allocB(2 * CN);
    __bf16* Htra   = allocB(2 * CN);
    float*  H5     = allocF(CN);
    float*  H6     = allocF(CN);

    auto cvt = [&](const float* src, __bf16* dst, size_t n) {
        cvt_bf16<<<(unsigned)((n + 255) / 256), 256, 0, stream>>>(src, dst, (int)n);
    };
    auto gemm = [&](const __bf16* A, const __bf16* Bmat, float* oF, __bf16* oB,
                    const float* bias, const float* add, const float* alpha,
                    int M_, int N_, int K_, int tA, int tB) {
        dim3 g(N_ / 128, M_ / 128);
        if (tA == 0 && tB == 0)
            gemm_k<0,0><<<g, 256, 0, stream>>>(A, Bmat, oF, oB, bias, add, alpha, M_, N_, K_);
        else if (tA == 1 && tB == 0)
            gemm_k<1,0><<<g, 256, 0, stream>>>(A, Bmat, oF, oB, bias, add, alpha, M_, N_, K_);
        else
            gemm_k<0,1><<<g, 256, 0, stream>>>(A, Bmat, oF, oB, bias, add, alpha, M_, N_, K_);
    };

    // ---- one-time conversions
    cvt(Wv,   Wv_b,   (size_t)C * C);
    cvt(Wv2,  Wv2_b,  (size_t)C * C);
    cvt(Wq,   Wq_b,   (size_t)C * C);
    cvt(Wk,   Wk_b,   (size_t)C * C);
    cvt(Wq2,  Wq2_b,  (size_t)C * C);
    cvt(Wk2,  Wk2_b,  (size_t)C * C);
    cvt(Win1, Win1_b, (size_t)C * C);
    cvt(Win3, Win3_b, (size_t)C * C);
    cvt(Wout2, Wo2_b, (size_t)C * C);
    cvt(Wout3, Wo3_b, (size_t)C * C);
    cvt(Wtr,  Wtr_b,  (size_t)(2 * C) * (2 * C));
    cvt(x, x_b, (size_t)NB * CN);
    cvt(y, y_b, (size_t)NB * CN);

    for (int b = 0; b < NB; ++b) {
        const float* xb = x + (size_t)b * CN;
        const __bf16* xbb = x_b + (size_t)b * CN;
        const __bf16* ybb = y_b + (size_t)b * CN;
        const float* zb = z + (size_t)b * C * 36;
        const float* mb = m + (size_t)b * C * 9;
        float* outb = out + (size_t)b * CN;

        // z/m branch (tiny)
        small_conv<<<(C * 36 + 255) / 256, 256, 0, stream>>>(Wv, zb, zv, C, 36);
        small_conv<<<(C * 9 + 255) / 256, 256, 0, stream>>>(Wv, mb, mv, C, 9);
        mline_k<<<(C * 36 + 255) / 256, 256, 0, stream>>>(mv, Wfc, bfc, ml);
        fusemz_k<<<(C * C + 255) / 256, 256, 0, stream>>>(zv, ml, fm, fm_b);

        // yv = Wv @ y
        gemm(Wv_b, ybb, yv, yv_b, nullptr, nullptr, nullptr, C, N, C, 0, 0);
        // fuse_mzy = fuse_mz @ yv + yv
        gemm(fm_b, yv_b, fmzy, fmzy_b, nullptr, yv, nullptr, C, N, C, 0, 0);
        // projections
        gemm(Wv2_b, xbb,    nullptr, v2_bf, nullptr, nullptr, nullptr, C, N, C, 0, 0);
        gemm(Wq_b,  xbb,    nullptr, q_bf,  bq,  nullptr, nullptr, C, N, C, 0, 0);
        gemm(Wk_b,  fmzy_b, nullptr, k_bf,  bk,  nullptr, nullptr, C, N, C, 0, 0);
        gemm(Wq2_b, xbb,    nullptr, q2_bf, bq2, nullptr, nullptr, C, N, C, 0, 0);
        gemm(Wk2_b, xbb,    nullptr, k2_bf, bk2, nullptr, nullptr, C, N, C, 0, 0);

        // branch 1: S = Q^T K ; softmax ; O^T = gamma * (P @ V^T)
        gemm(q_bf, k_bf, S, nullptr, nullptr, nullptr, nullptr, N, N, C, 1, 0);
        softmax_rows<<<N, 256, 0, stream>>>(S, P, N);
        gemm(P, fmzy_b, o1T, o1T_b, nullptr, nullptr, gamma, N, C, N, 0, 1);

        // branch 2
        gemm(q2_bf, k2_bf, S, nullptr, nullptr, nullptr, nullptr, N, N, C, 1, 0);
        softmax_rows<<<N, 256, 0, stream>>>(S, P, N);
        gemm(P, v2_bf, o2T, o2T_b, nullptr, nullptr, gamma2, N, C, N, 0, 1);

        // gated fusion
        gemm(Win1_b, o1T_b, nullptr, Hcat,       bin1, nullptr, nullptr, C, N, C, 0, 1);
        gemm(Win3_b, o2T_b, nullptr, Hcat + CN,  bin3, nullptr, nullptr, C, N, C, 0, 1);
        gemm(Wtr_b, Hcat, nullptr, Htra, btr, nullptr, nullptr, 2 * C, N, 2 * C, 0, 0);
        gemm(Wo2_b, Htra,      H5, nullptr, bout2, nullptr, nullptr, C, N, C, 0, 0);
        gemm(Wo3_b, Htra + CN, H6, nullptr, bout3, nullptr, nullptr, C, N, C, 0, 0);

        final_k<<<(unsigned)((CN + 255) / 256), 256, 0, stream>>>(
            H5, H6, o1T, o2T, xb, outb, C, N);
    }
}
